// GAGNNUpdate_83743272337785
// MI455X (gfx1250) — matrix operand order repaired and checked by reading, not compiled, not run on hardware
//
#include <hip/hip_runtime.h>
#include <hip/hip_bf16.h>

// ---------------- problem constants ----------------
constexpr int Bb = 64, Cc = 400, Gg = 16, Tt = 8, Dd = 64;
constexpr int Ee = 6400, PREDn = 24;
constexpr int XEM = 32, LOCEM = 12, DATEEM = 4, EDGEH = 12, GNNH = 32, DFFn = 256, NH = 4;
constexpr int Rr = Bb * Cc * Tt;      // 204800 encoder rows
constexpr int RC = Bb * Cc;           // 25600 city-nodes
constexpr int EG = Gg * (Gg - 1);     // 240 group pairs per batch
constexpr int EGB = Bb * EG;          // 15360 group edges
constexpr int EBIG = Bb * Ee;         // 409600 city edges
constexpr int FFC = Rr / 4;           // 51200-row FFN chunk

typedef __attribute__((ext_vector_type(16))) _Float16 v16h;
typedef __attribute__((ext_vector_type(8)))  _Float16 v8h;
typedef __attribute__((ext_vector_type(8)))  float    v8f;
typedef __attribute__((ext_vector_type(4)))  float    v4f;

// -------- weight pack: W[K,N] f32 -> P[(K/32)][N][32] f16 (WMMA B-fragment order) --------
__global__ void pack_w_kernel(const float* __restrict__ W, _Float16* __restrict__ P,
                              int K, int N) {
  int idx = blockIdx.x * blockDim.x + threadIdx.x;
  if (idx >= K * N) return;
  int n = idx % N, k = idx / N;
  int kb = k >> 5, kr = k & 31;
  P[((size_t)kb * N + n) * 32 + kr] = (_Float16)W[(size_t)k * N + n];
}

// ---------------- WMMA GEMM: C[M,N] = relu?(A[M,K] @ W[K,N] + bias) ----------------
// One wave owns a 16 x (16*NT) tile of C; K stepped by 32 via v_wmma_f32_16x16x32_f16.
// Fragment layouts per CDNA5 ISA 7.12.2 (wave32):
//   A: lane m=l&15, K = e + (e>=8?8:0) + (l>=16?8:0)   -> 4x global_load_b128 (f32)
//   B: lane n=l&15, K = e + (e>=8?8:0) + (l>=16?8:0)               -> 2x global_load_b128 (packed f16)
//   C: lane n=l&15, row = e + (l>=16?8:0)
template <int NT>
__global__ void gemm_wmma_kernel(const float* __restrict__ A,
                                 const _Float16* __restrict__ Wp,
                                 const float* __restrict__ bias,
                                 float* __restrict__ Cout,
                                 int M, int K, int N, int relu) {
  int wave = (blockIdx.x * blockDim.x + threadIdx.x) >> 5;
  int lane = threadIdx.x & 31;
  int tilesM = M >> 4;
  int tilesN = N / (16 * NT);
  int tm = wave % tilesM;
  int tn = wave / tilesM;
  if (tn >= tilesN) return;                       // wave-uniform: EXEC stays all-1 for WMMA
  int m      = (tm << 4) + (lane & 15);
  int kaBase = (lane >> 4) << 3;                  // 0 / 8
  int kbBase = (lane >> 4) << 3;                  // 0 / 8
  int nBase  = tn * NT * 16 + (lane & 15);
  v8f zero = {};
  v8f acc[NT];
#pragma unroll
  for (int t = 0; t < NT; ++t) acc[t] = zero;

  for (int k0 = 0; k0 < K; k0 += 32) {
    const v4f* arow4 = (const v4f*)(A + (size_t)m * K + k0 + kaBase);
    if (k0 + 32 < K) __builtin_prefetch((const float*)arow4 + 32, 0, 1);
    v4f a0 = arow4[0], a1 = arow4[1], a2 = arow4[4], a3 = arow4[5];
    v16h afr;
#pragma unroll
    for (int e = 0; e < 4; ++e) {
      afr[e]      = (_Float16)a0[e];
      afr[4 + e]  = (_Float16)a1[e];
      afr[8 + e]  = (_Float16)a2[e];
      afr[12 + e] = (_Float16)a3[e];
    }
    int kb = k0 >> 5;
#pragma unroll
    for (int t = 0; t < NT; ++t) {
      int n = nBase + t * 16;
      const v8h* wp = (const v8h*)(Wp + ((size_t)kb * N + n) * 32 + kbBase);
      v8h lo = wp[0], hi = wp[2];
      v16h bfr = __builtin_shufflevector(lo, hi, 0, 1, 2, 3, 4, 5, 6, 7,
                                         8, 9, 10, 11, 12, 13, 14, 15);
      acc[t] = __builtin_amdgcn_wmma_f32_16x16x32_f16(
          false, afr, false, bfr, (short)0, acc[t], false, false);
    }
  }
  int mrow = (tm << 4) + ((lane >> 4) << 3);
#pragma unroll
  for (int t = 0; t < NT; ++t) {
    int n = nBase + t * 16;
    float bn = bias ? bias[n] : 0.f;
#pragma unroll
    for (int e = 0; e < 8; ++e) {
      float v = acc[t][e] + bn;
      if (relu) v = v > 0.f ? v : 0.f;
      Cout[(size_t)(mrow + e) * N + n] = v;
    }
  }
}

// ---------------- residual + LayerNorm over rows of 64 (wave per row) ----------------
__global__ void add_ln_kernel(const float* __restrict__ a, const float* __restrict__ b,
                              const float* __restrict__ g, const float* __restrict__ be,
                              float* __restrict__ out, int Mrows) {
  int wave = (blockIdx.x * blockDim.x + threadIdx.x) >> 5;
  int lane = threadIdx.x & 31;
  if (wave >= Mrows) return;
  size_t base = (size_t)wave * 64;
  float v0 = a[base + lane] + b[base + lane];
  float v1 = a[base + 32 + lane] + b[base + 32 + lane];
  float s = v0 + v1;
  for (int o = 16; o > 0; o >>= 1) s += __shfl_xor(s, o);
  float mean = s * (1.f / 64.f);
  float d0 = v0 - mean, d1 = v1 - mean;
  float q = d0 * d0 + d1 * d1;
  for (int o = 16; o > 0; o >>= 1) q += __shfl_xor(q, o);
  float inv = rsqrtf(q * (1.f / 64.f) + 1e-5f);
  out[base + lane]      = d0 * inv * g[lane] + be[lane];
  out[base + 32 + lane] = d1 * inv * g[lane + 32] + be[lane + 32];
}

// ---------------- attention: T=8, 4 heads of 16; thread per (seq,head,t) ----------------
__global__ void attn_kernel(const float* __restrict__ Q, const float* __restrict__ Km,
                            const float* __restrict__ V, float* __restrict__ O) {
  int idx = blockIdx.x * blockDim.x + threadIdx.x;
  if (idx >= RC * NH * Tt) return;
  int t = idx & 7, h = (idx >> 3) & 3, seq = idx >> 5;
  const float* qp = Q + ((size_t)(seq * 8 + t) * 64 + h * 16);
  float sc[8], mx = -1e30f;
  for (int s = 0; s < 8; ++s) {
    const float* kp = Km + ((size_t)(seq * 8 + s) * 64 + h * 16);
    float d = 0.f;
    for (int j = 0; j < 16; ++j) d += qp[j] * kp[j];
    d *= 0.25f;                                   // 1/sqrt(16)
    sc[s] = d; mx = fmaxf(mx, d);
  }
  float sum = 0.f;
  for (int s = 0; s < 8; ++s) { sc[s] = __expf(sc[s] - mx); sum += sc[s]; }
  float inv = 1.f / sum;
  for (int j = 0; j < 16; ++j) {
    float o = 0.f;
    for (int s = 0; s < 8; ++s)
      o += sc[s] * V[(size_t)(seq * 8 + s) * 64 + h * 16 + j];
    O[(size_t)(seq * 8 + t) * 64 + h * 16 + j] = o * inv;
  }
}

// ---------------- tiny elementwise / setup kernels ----------------
__global__ void zero_kernel(float* __restrict__ p, size_t n) {
  size_t i = (size_t)blockIdx.x * blockDim.x + threadIdx.x;
  if (i < n) p[i] = 0.f;
}
__global__ void softmax_w_kernel(const float* __restrict__ w, float* __restrict__ wsm) {
  int c = blockIdx.x * blockDim.x + threadIdx.x;
  if (c >= Cc) return;
  float mx = -1e30f;
  for (int g = 0; g < Gg; ++g) mx = fmaxf(mx, w[c * Gg + g]);
  float ex[Gg], sum = 0.f;
  for (int g = 0; g < Gg; ++g) { ex[g] = __expf(w[c * Gg + g] - mx); sum += ex[g]; }
  float inv = 1.f / sum;
  for (int g = 0; g < Gg; ++g) wsm[c * Gg + g] = ex[g] * inv;
}
__global__ void loc_kernel(const float* __restrict__ city, const float* __restrict__ w,
                           const float* __restrict__ b, float* __restrict__ loc) {
  int idx = blockIdx.x * blockDim.x + threadIdx.x;
  if (idx >= Cc * LOCEM) return;
  int c = idx / LOCEM, f = idx % LOCEM;
  loc[idx] = b[f] + city[c * 2] * w[f] + city[c * 2 + 1] * w[LOCEM + f];
}
__global__ void uem_kernel(const int* __restrict__ u, const float* __restrict__ e1,
                           const float* __restrict__ e2, const float* __restrict__ e3,
                           float* __restrict__ uem) {
  int idx = blockIdx.x * blockDim.x + threadIdx.x;
  if (idx >= Bb * 12) return;
  int b = idx / 12, k = idx % 12, which = k >> 2, sub = k & 3;
  int id = u[b * 3 + which];
  const float* tab = which == 0 ? e1 : (which == 1 ? e2 : e3);
  uem[idx] = tab[id * 4 + sub];
}
// g_x[b,g,f] = sum_c wsm[c,g] * (f<32 ? h[b,c,f] : loc[c,f-32])
__global__ void gx_enc_kernel(const float* __restrict__ wsm, const float* __restrict__ h,
                              const float* __restrict__ loc, float* __restrict__ gx) {
  int idx = blockIdx.x * blockDim.x + threadIdx.x;
  if (idx >= Bb * Gg * 44) return;
  int f = idx % 44, g = (idx / 44) % Gg, b = idx / (44 * Gg);
  float s = 0.f;
  if (f < 32)
    for (int c = 0; c < Cc; ++c) s += wsm[c * Gg + g] * h[((size_t)b * Cc + c) * 32 + f];
  else
    for (int c = 0; c < Cc; ++c) s += wsm[c * Gg + g] * loc[c * LOCEM + (f - 32)];
  gx[idx] = s;
}
// decoder: dg[b,g,f] = sum_c w_raw[c,g]*xin[b,c,f]   (F = 32)
__global__ void group_agg_kernel(const float* __restrict__ wmat, const float* __restrict__ xin,
                                 float* __restrict__ outg, int F) {
  int idx = blockIdx.x * blockDim.x + threadIdx.x;
  if (idx >= Bb * Gg * F) return;
  int f = idx % F, g = (idx / F) % Gg, b = idx / (F * Gg);
  float s = 0.f;
  for (int c = 0; c < Cc; ++c) s += wmat[c * Gg + g] * xin[((size_t)b * Cc + c) * F + f];
  outg[idx] = s;
}
// out[b,c, outOff+f] = sum_g wmat[c,g]*gin[b,g,f]
__global__ void group_mix_kernel(const float* __restrict__ wmat, const float* __restrict__ gin,
                                 float* __restrict__ outp, int F, int outStride, int outOff) {
  int idx = blockIdx.x * blockDim.x + threadIdx.x;
  if (idx >= Bb * Cc * F) return;
  int f = idx % F, c = (idx / F) % Cc, b = idx / (F * Cc);
  float s = 0.f;
#pragma unroll
  for (int g = 0; g < Gg; ++g) s += wmat[c * Gg + g] * gin[((size_t)b * Gg + g) * F + f];
  outp[(size_t)(b * Cc + c) * outStride + outOff + f] = s;
}
__global__ void strided_copy_kernel(const float* __restrict__ src, float* __restrict__ dst,
                                    int F, int outStride, int outOff, int total) {
  int i = blockIdx.x * blockDim.x + threadIdx.x;
  if (i >= total) return;
  int n = i / F, f = i % F;
  dst[(size_t)n * outStride + outOff + f] = src[i];
}
__global__ void build_gidx_kernel(int* __restrict__ grow, int* __restrict__ gcol) {
  int idx = blockIdx.x * blockDim.x + threadIdx.x;
  if (idx >= EGB) return;
  int b = idx / EG, p = idx % EG;
  int i = p / 15, jr = p % 15;
  int j = jr + (jr >= i ? 1 : 0);                 // ordered (i,j), i != j, loop order
  grow[idx] = b * Gg + i;
  gcol[idx] = b * Gg + j;
}
__global__ void build_bidx_kernel(const int* __restrict__ ei, int* __restrict__ brow,
                                  int* __restrict__ bcol) {
  int idx = blockIdx.x * blockDim.x + threadIdx.x;
  if (idx >= EBIG) return;
  int b = idx / Ee, e = idx % Ee;
  brow[idx] = ei[((size_t)b * 2 + 0) * Ee + e] + b * Cc;
  bcol[idx] = ei[((size_t)b * 2 + 1) * Ee + e] + b * Cc;
}

// ---------------- edge_inf: 100 -> 12, wave per group edge, LDS-staged ----------------
__global__ void edge_inf_kernel(const float* __restrict__ gx, const float* __restrict__ uem,
                                const int* __restrict__ grow, const int* __restrict__ gcol,
                                const float* __restrict__ w, const float* __restrict__ b,
                                float* __restrict__ gew) {
  __shared__ float sm[4 * 104];
  int lane = threadIdx.x & 31, wly = threadIdx.x >> 5;
  int e = (blockIdx.x * blockDim.x + threadIdx.x) >> 5;
  float* s = sm + wly * 104;
  if (e < EGB) {
    int bi = e / EG, src = grow[e], dst = gcol[e];
    for (int i = lane; i < 44; i += 32) {
      s[i]      = gx[(size_t)src * 44 + i];
      s[44 + i] = gx[(size_t)dst * 44 + i];
    }
    if (lane < 12) s[88 + lane] = uem[bi * 12 + lane];
  }
  __syncthreads();
  if (e < EGB && lane < EDGEH) {
    float o = b[lane];
    for (int i = 0; i < 100; ++i) o += s[i] * w[i * EDGEH + lane];
    gew[(size_t)e * EDGEH + lane] = o > 0.f ? o : 0.f;
  }
}

// ------- node-model edge MLP fused with scatter-add (wave per edge, lane = out dim) -------
__global__ void edge_mlp_scatter_kernel(const float* __restrict__ xn, int F1,
                                        const float* __restrict__ ea, int F2,
                                        const int* __restrict__ row, const int* __restrict__ col,
                                        const float* __restrict__ w, const float* __restrict__ b,
                                        float* __restrict__ agg, float* __restrict__ cnt, int E) {
  __shared__ float sm[4 * 112];
  int lane = threadIdx.x & 31, wly = threadIdx.x >> 5;
  int e = (blockIdx.x * blockDim.x + threadIdx.x) >> 5;
  float* s = sm + wly * 112;
  int Fin = F1 + F2;
  if (e < E) {
    int r = row[e];
    for (int i = lane; i < F1; i += 32) s[i] = xn[(size_t)r * F1 + i];
    for (int i = lane; i < F2; i += 32) s[F1 + i] = ea[(size_t)e * F2 + i];
  }
  __syncthreads();
  if (e < E) {
    float o = b[lane];
    for (int i = 0; i < Fin; ++i) o += s[i] * w[i * 32 + lane];
    o = o > 0.f ? o : 0.f;
    int c = col[e];
    atomicAdd(&agg[(size_t)c * 32 + lane], o);
    if (lane == 0) atomicAdd(&cnt[c], 1.0f);
  }
}

// ------- node MLP: relu(W2 @ concat(x, agg/max(cnt,1)) + b2), wave per node -------
__global__ void node_mlp_kernel(const float* __restrict__ xn, int F1,
                                const float* __restrict__ agg, const float* __restrict__ cnt,
                                const float* __restrict__ w, const float* __restrict__ b,
                                float* __restrict__ out, int Nn) {
  __shared__ float sm[4 * 112];
  int lane = threadIdx.x & 31, wly = threadIdx.x >> 5;
  int n = (blockIdx.x * blockDim.x + threadIdx.x) >> 5;
  float* s = sm + wly * 112;
  int Fin = F1 + 32;
  if (n < Nn) {
    for (int i = lane; i < F1; i += 32) s[i] = xn[(size_t)n * F1 + i];
    float invc = 1.f / fmaxf(cnt[n], 1.f);
    s[F1 + lane] = agg[(size_t)n * 32 + lane] * invc;
  }
  __syncthreads();
  if (n < Nn) {
    float o = b[lane];
    for (int i = 0; i < Fin; ++i) o += s[i] * w[i * 32 + lane];
    out[(size_t)n * 32 + lane] = o > 0.f ? o : 0.f;
  }
}

// ---------------- fused prediction head: 32 ->relu 16 ->relu 24 ----------------
__global__ void pred_kernel(const float* __restrict__ xn,
                            const float* __restrict__ w1, const float* __restrict__ b1,
                            const float* __restrict__ w2, const float* __restrict__ b2,
                            float* __restrict__ out) {
  __shared__ float sm[4 * 48];
  int lane = threadIdx.x & 31, wly = threadIdx.x >> 5;
  int n = (blockIdx.x * blockDim.x + threadIdx.x) >> 5;
  float* s = sm + wly * 48;
  if (n < RC) s[lane] = xn[(size_t)n * 32 + lane];
  __syncthreads();
  if (n < RC && lane < 16) {
    float o = b1[lane];
    for (int i = 0; i < 32; ++i) o += s[i] * w1[i * 16 + lane];
    s[32 + lane] = o > 0.f ? o : 0.f;
  }
  __syncthreads();
  if (n < RC && lane < PREDn) {
    float o = b2[lane];
    for (int i = 0; i < 16; ++i) o += s[32 + i] * w2[i * PREDn + lane];
    out[(size_t)n * PREDn + lane] = o > 0.f ? o : 0.f;
  }
}

// ---------------- host orchestration ----------------
extern "C" void kernel_launch(void* const* d_in, const int* in_sizes, int n_in,
                              void* d_out, int out_size, void* d_ws, size_t ws_size,
                              hipStream_t stream) {
  // inputs (recursive dict-insertion-order flatten of setup_inputs())
  const float* x      = (const float*)d_in[0];
  const int*   eidx   = (const int*)d_in[1];
  const float* edgeW  = (const float*)d_in[2];
  const int*   u      = (const int*)d_in[3];
  const float* city   = (const float*)d_in[4];
  const float *wq_w=(const float*)d_in[5],  *wq_b=(const float*)d_in[6];
  const float *wk_w=(const float*)d_in[7],  *wk_b=(const float*)d_in[8];
  const float *wv_w=(const float*)d_in[9],  *wv_b=(const float*)d_in[10];
  const float *wo_w=(const float*)d_in[11], *wo_b=(const float*)d_in[12];
  const float *ln1g=(const float*)d_in[13], *ln1b=(const float*)d_in[14];
  const float *ff1w=(const float*)d_in[15], *ff1b=(const float*)d_in[16];
  const float *ff2w=(const float*)d_in[17], *ff2b=(const float*)d_in[18];
  const float *ln2g=(const float*)d_in[19], *ln2b=(const float*)d_in[20];
  const float *w_raw=(const float*)d_in[21];
  const float *xew=(const float*)d_in[22],  *xeb=(const float*)d_in[23];
  const float *locw=(const float*)d_in[24], *locb=(const float*)d_in[25];
  const float *emb1=(const float*)d_in[26], *emb2=(const float*)d_in[27], *emb3=(const float*)d_in[28];
  const float *eiw=(const float*)d_in[29],  *eib=(const float*)d_in[30];
  const float *g0m1w=(const float*)d_in[31],*g0m1b=(const float*)d_in[32];
  const float *g0m2w=(const float*)d_in[33],*g0m2b=(const float*)d_in[34];
  const float *g1m1w=(const float*)d_in[35],*g1m1b=(const float*)d_in[36];
  const float *g1m2w=(const float*)d_in[37],*g1m2b=(const float*)d_in[38];
  const float *q0m1w=(const float*)d_in[39],*q0m1b=(const float*)d_in[40];
  const float *q0m2w=(const float*)d_in[41],*q0m2b=(const float*)d_in[42];
  const float *q1m1w=(const float*)d_in[43],*q1m1b=(const float*)d_in[44];
  const float *q1m2w=(const float*)d_in[45],*q1m2b=(const float*)d_in[46];
  const float *dxew=(const float*)d_in[47], *dxeb=(const float*)d_in[48];
  const float *e0m1w=(const float*)d_in[49],*e0m1b=(const float*)d_in[50];
  const float *e0m2w=(const float*)d_in[51],*e0m2b=(const float*)d_in[52];
  const float *e1m1w=(const float*)d_in[53],*e1m1b=(const float*)d_in[54];
  const float *e1m2w=(const float*)d_in[55],*e1m2b=(const float*)d_in[56];
  const float *f0m1w=(const float*)d_in[57],*f0m1b=(const float*)d_in[58];
  const float *f0m2w=(const float*)d_in[59],*f0m2b=(const float*)d_in[60];
  const float *f1m1w=(const float*)d_in[61],*f1m1b=(const float*)d_in[62];
  const float *f1m2w=(const float*)d_in[63],*f1m2b=(const float*)d_in[64];
  const float *p1w=(const float*)d_in[65],  *p1b=(const float*)d_in[66];
  const float *p2w=(const float*)d_in[67],  *p2b=(const float*)d_in[68];

  // workspace carve-up (float-granular, 256B aligned)
  float* wsf = (float*)d_ws;
  size_t off = 0;
  auto alloc = [&](size_t n) { float* p = wsf + off; off += (n + 63) & ~(size_t)63; return p; };
  float* bq   = alloc((size_t)Rr * 64);
  float* bk   = alloc((size_t)Rr * 64);
  float* bv   = alloc((size_t)Rr * 64);
  float* bt1  = alloc((size_t)Rr * 64);
  float* bx   = alloc((size_t)Rr * 64);
  float* bff  = alloc((size_t)FFC * 256);
  float* bh   = alloc((size_t)RC * 32);
  float* bwsm = alloc(Cc * Gg);
  float* bloc = alloc(Cc * LOCEM);
  float* bgx  = alloc((size_t)Bb * Gg * 44);
  float* buem = alloc(Bb * 12);
  float* bgew = alloc((size_t)EGB * EDGEH);
  float* bagg = alloc((size_t)RC * 32);
  float* bcnt = alloc(RC);
  float* bgxa = alloc((size_t)Bb * Gg * 32);
  float* bgxb = alloc((size_t)Bb * Gg * 32);
  float* bnx64= alloc((size_t)RC * 64);
  float* bnxa = alloc((size_t)RC * 32);
  float* bnxb = alloc((size_t)RC * 32);
  float* bdx  = alloc((size_t)RC * 32);
  int* grow = (int*)alloc(EGB);
  int* gcol = (int*)alloc(EGB);
  int* brow = (int*)alloc(EBIG);
  int* bcol = (int*)alloc(EBIG);
  // packed f16 weights (allocated in float units; 1 float = 2 halves)
  _Float16* pq   = (_Float16*)alloc(64 * 64 / 2);
  _Float16* pk   = (_Float16*)alloc(64 * 64 / 2);
  _Float16* pv   = (_Float16*)alloc(64 * 64 / 2);
  _Float16* po   = (_Float16*)alloc(64 * 64 / 2);
  _Float16* pff1 = (_Float16*)alloc(64 * 256 / 2);
  _Float16* pff2 = (_Float16*)alloc(256 * 64 / 2);
  _Float16* pxe  = (_Float16*)alloc(512 * 32 / 2);
  _Float16* pdxe = (_Float16*)alloc(32 * 32 / 2);

  auto cdiv = [](int a, int b) { return (a + b - 1) / b; };
  auto pack = [&](const float* W, _Float16* P, int K, int N) {
    pack_w_kernel<<<cdiv(K * N, 256), 256, 0, stream>>>(W, P, K, N);
  };
  auto gemm = [&](int NT, const float* A, const _Float16* Wp, const float* bias, float* C,
                  int M, int K, int N, int relu) {
    int waves = (M / 16) * (N / (16 * NT));
    int grid = cdiv(waves * 32, 256);
    if (NT == 4) gemm_wmma_kernel<4><<<grid, 256, 0, stream>>>(A, Wp, bias, C, M, K, N, relu);
    else         gemm_wmma_kernel<2><<<grid, 256, 0, stream>>>(A, Wp, bias, C, M, K, N, relu);
  };
  auto zero = [&](float* p, size_t n) {
    zero_kernel<<<cdiv((int)n, 256), 256, 0, stream>>>(p, n);
  };

  // one-shot prep: edge indices + weight packing
  build_gidx_kernel<<<cdiv(EGB, 256), 256, 0, stream>>>(grow, gcol);
  build_bidx_kernel<<<cdiv(EBIG, 256), 256, 0, stream>>>(eidx, brow, bcol);
  pack(wq_w, pq, 64, 64);   pack(wk_w, pk, 64, 64);
  pack(wv_w, pv, 64, 64);   pack(wo_w, po, 64, 64);
  pack(ff1w, pff1, 64, 256); pack(ff2w, pff2, 256, 64);
  pack(xew, pxe, 512, 32);   pack(dxew, pdxe, 32, 32);

  // ---- transformer encoder (WMMA GEMMs + VALU attention + LN) ----
  gemm(4, x, pq, wq_b, bq, Rr, 64, 64, 0);
  gemm(4, x, pk, wk_b, bk, Rr, 64, 64, 0);
  gemm(4, x, pv, wv_b, bv, Rr, 64, 64, 0);
  attn_kernel<<<cdiv(RC * NH * Tt, 256), 256, 0, stream>>>(bq, bk, bv, bt1);
  gemm(4, bt1, po, wo_b, bk, Rr, 64, 64, 0);                   // o-proj -> bk (K dead)
  add_ln_kernel<<<cdiv(Rr * 32, 256), 256, 0, stream>>>(x, bk, ln1g, ln1b, bx, Rr);
  for (int c = 0; c < 4; ++c) {                                 // FFN, chunked x4
    gemm(4, bx + (size_t)c * FFC * 64, pff1, ff1b, bff, FFC, 64, 256, 1);
    gemm(4, bff, pff2, ff2b, bt1 + (size_t)c * FFC * 64, FFC, 256, 64, 0);
  }
  add_ln_kernel<<<cdiv(Rr * 32, 256), 256, 0, stream>>>(bx, bt1, ln2g, ln2b, bx, Rr);

  // ---- embeddings / group graph features ----
  gemm(2, bx, pxe, xeb, bh, RC, 512, 32, 0);                    // x_embed: [25600,512]@[512,32]
  softmax_w_kernel<<<cdiv(Cc, 128), 128, 0, stream>>>(w_raw, bwsm);
  loc_kernel<<<cdiv(Cc * LOCEM, 256), 256, 0, stream>>>(city, locw, locb, bloc);
  gx_enc_kernel<<<cdiv(Bb * Gg * 44, 256), 256, 0, stream>>>(bwsm, bh, bloc, bgx);
  uem_kernel<<<cdiv(Bb * 12, 128), 128, 0, stream>>>(u, emb1, emb2, emb3, buem);
  edge_inf_kernel<<<cdiv(EGB * 32, 128), 128, 0, stream>>>(bgx, buem, grow, gcol, eiw, eib, bgew);

  const int NG = Bb * Gg;
  auto node_model = [&](const float* xn, int F1, const float* ea, int F2,
                        const int* row, const int* col, int E, int Nn,
                        const float* m1w, const float* m1b,
                        const float* m2w, const float* m2b, float* out) {
    zero(bagg, (size_t)Nn * 32);
    zero(bcnt, Nn);
    edge_mlp_scatter_kernel<<<cdiv(E * 32, 128), 128, 0, stream>>>(
        xn, F1, ea, F2, row, col, m1w, m1b, bagg, bcnt, E);
    node_mlp_kernel<<<cdiv(Nn * 32, 128), 128, 0, stream>>>(
        xn, F1, bagg, bcnt, m2w, m2b, out, Nn);
  };

  // ---- encoder GNN ----
  node_model(bgx, 44, bgew, EDGEH, grow, gcol, EGB, NG, g0m1w, g0m1b, g0m2w, g0m2b, bgxa);
  node_model(bgxa, 32, bgew, EDGEH, grow, gcol, EGB, NG, g1m1w, g1m1b, g1m2w, g1m2b, bgxb);
  strided_copy_kernel<<<cdiv(RC * 32, 256), 256, 0, stream>>>(bh, bnx64, 32, 64, 0, RC * 32);
  group_mix_kernel<<<cdiv(Bb * Cc * 32, 256), 256, 0, stream>>>(bwsm, bgxb, bnx64, 32, 64, 32);
  node_model(bnx64, 64, edgeW, 1, brow, bcol, EBIG, RC, q0m1w, q0m1b, q0m2w, q0m2b, bnxa);
  node_model(bnxa, 32, edgeW, 1, brow, bcol, EBIG, RC, q1m1w, q1m1b, q1m2w, q1m2b, bnxb);

  // ---- decoder (raw w) ----
  gemm(2, bnxb, pdxe, dxeb, bdx, RC, 32, 32, 0);                // dec.x_embed (WMMA, K=32)
  group_agg_kernel<<<cdiv(Bb * Gg * 32, 256), 256, 0, stream>>>(w_raw, bdx, bgxa, 32);
  node_model(bgxa, 32, bgew, EDGEH, grow, gcol, EGB, NG, e0m1w, e0m1b, e0m2w, e0m2b, bgxb);
  node_model(bgxb, 32, bgew, EDGEH, grow, gcol, EGB, NG, e1m1w, e1m1b, e1m2w, e1m2b, bgxa);
  strided_copy_kernel<<<cdiv(RC * 32, 256), 256, 0, stream>>>(bdx, bnx64, 32, 64, 0, RC * 32);
  group_mix_kernel<<<cdiv(Bb * Cc * 32, 256), 256, 0, stream>>>(w_raw, bgxa, bnx64, 32, 64, 32);
  node_model(bnx64, 64, edgeW, 1, brow, bcol, EBIG, RC, f0m1w, f0m1b, f0m2w, f0m2b, bnxa);
  node_model(bnxa, 32, edgeW, 1, brow, bcol, EBIG, RC, f1m1w, f1m1b, f1m2w, f1m2b, bdx);

  // ---- prediction head ----
  pred_kernel<<<cdiv(RC * 32, 128), 128, 0, stream>>>(bdx, p1w, p1b, p2w, p2b, (float*)d_out);
}
